// MaskUnitAttention_412316860487
// MI455X (gfx1250) — matrix operand order repaired and checked by reading, not compiled, not run on hardware
//
#include <hip/hip_runtime.h>

typedef __attribute__((ext_vector_type(16))) _Float16 v16h;
typedef __attribute__((ext_vector_type(8)))  float    v8f;
typedef __attribute__((ext_vector_type(4)))  unsigned int u32x4;
typedef __attribute__((ext_vector_type(8)))  unsigned int u32x8;
typedef _Float16 f16;

#define DIMIN   96
#define DIM_OUT 192
#define NHEADS  2
#define HEAD_D  96
#define WIN     16      // window size after q pooling
#define TPW     64      // tokens per window (Q_STRIDE * WINDOW_SIZE)
#define NW      49      // number of windows
#define NTOK    3136
#define NBATCH  64
#define QKV_C   576     // 3 * DIM_OUT

// ---- dynamic LDS layout (bytes) ----
// xs    f16 [64][96]        @ 0       (12288)
// qkvs  f16 [64][576]       @ 12288   (73728)
// qp    f16 [2][16][96]     @ 86016   (6144)   pooled Q, pre-scaled
// S     f32 [2][16][64]     @ 92160   (8192)   attention logits
// P     f16 [2][16][64]     @ 100352  (4096)   softmax probs
// Ot    f16 [16][192]       @ 104448  (6144)   attn output (heads concat)
// xsraw f32 [64][96]        @ 110592  (24576)  TDM staging (raw f32 tile)
#define XSRAW_OFF  110592
#define SMEM_BYTES 135168

// A-matrix (16x32 f16) K pattern: VGPR v, lane-half h -> first K of the f16 pair
__device__ __forceinline__ int kpatA(int v, int half) {
    return ((v >> 2) << 4) + ((v & 3) << 1) + (half << 3);
}
// B-matrix (32x16 f16) K pattern
__device__ __forceinline__ int kpatB(int v, int half) {
    return ((v >> 2) << 4) + ((v & 3) << 1) + (half << 3);
}

__global__ void __launch_bounds__(256)
muattn_fused_kernel(const float* __restrict__ x,
                    const float* __restrict__ w_qkv, const float* __restrict__ b_qkv,
                    const float* __restrict__ w_proj, const float* __restrict__ b_proj,
                    float* __restrict__ out)
{
    extern __shared__ char smem[];
    f16*   xs    = (f16*)(smem);
    f16*   qkvs  = (f16*)(smem + 12288);
    f16*   qp    = (f16*)(smem + 86016);
    float* S     = (float*)(smem + 92160);
    f16*   P     = (f16*)(smem + 100352);
    f16*   Ot    = (f16*)(smem + 104448);
    float* xsraw = (float*)(smem + XSRAW_OFF);

    const int tid  = threadIdx.x;
    const int lane = tid & 31;
    const int wv   = tid >> 5;       // wave 0..7
    const int l16  = lane & 15;
    const int half = lane >> 4;

    const int b    = blockIdx.x / NW;
    const int wwin = blockIdx.x % NW;

    const float* xb = x + (size_t)b * NTOK * DIMIN;

    // ---- warm L2/WGP$ for the shared weight matrices (global_prefetch_b8)
    {
        const char* wq = (const char*)w_qkv;
        for (unsigned o = (unsigned)tid * 256u; o < QKV_C * DIMIN * 4u; o += 256u * 256u)
            __builtin_prefetch(wq + o, 0, 3);
        const char* wp = (const char*)w_proj;
        for (unsigned o = (unsigned)tid * 256u; o < DIM_OUT * DIM_OUT * 4u; o += 256u * 256u)
            __builtin_prefetch(wp + o, 0, 3);
    }

    // ---- Phase A: TDM async tensor-load of the 64 strided window rows into LDS.
    // 2D tile: 96 f32 per row, 64 rows, row stride 49*96 elements.
    if (wv == 0) {
        unsigned long long gaddr = (unsigned long long)(const void*)(xb + (size_t)wwin * DIMIN);
        unsigned int ldsoff = (unsigned int)(unsigned long long)(smem + XSRAW_OFF);
        u32x4 g0;
        g0.x = 1u;                                        // count=1 (valid user D#)
        g0.y = ldsoff;                                    // lds_addr (bytes)
        g0.z = (unsigned int)gaddr;                       // global_addr[31:0]
        g0.w = ((unsigned int)(gaddr >> 32) & 0x01FFFFFFu)// global_addr[56:32]
             | 0x80000000u;                               // type=2 ("image")
        u32x8 g1;
        g1.s0 = 0x00020000u;                              // wg_mask=0, data_size=2 (4B)
        g1.s1 = ((unsigned int)DIMIN) << 16;              // tensor_dim0[15:0]=96
        g1.s2 = ((unsigned int)TPW)   << 16;              // tensor_dim0[31:16]=0 | tensor_dim1[15:0]=64
        g1.s3 = ((unsigned int)DIMIN) << 16;              // tensor_dim1[31:16]=0 | tile_dim0=96
        g1.s4 = (unsigned int)TPW;                        // tile_dim1=64, tile_dim2=0
        g1.s5 = (unsigned int)(NW * DIMIN);               // tensor_dim0_stride[31:0]=4704
        g1.s6 = 0u;                                       // stride hi / dim1_stride lo
        g1.s7 = 0u;                                       // dim1_stride hi (unused, 2D tile)
        asm volatile("tensor_load_to_lds %0, %1" :: "s"(g0), "s"(g1) : "memory");
        __builtin_amdgcn_s_wait_tensorcnt(0);
    }
    __syncthreads();

    // convert staged f32 tile -> f16 working copy
    for (int i = tid; i < TPW * DIMIN; i += 256)
        xs[i] = (f16)xsraw[i];
    __syncthreads();

    // ---- Phase B: QKV = xs[64x96] @ w_qkv^T[96x576] + b_qkv  -> qkvs f16 [64][576]
    // nt-outer: convert each weight B-fragment once, reuse across the 4 m-tiles.
    for (int nt = wv; nt < 36; nt += 8) {
        const int n = nt * 16 + l16;                      // output channel == row of w_qkv
        v16h bfrag[3];
#pragma unroll
        for (int ks = 0; ks < 3; ++ks) {
#pragma unroll
            for (int v = 0; v < 8; ++v) {
                int kb = ks * 32 + kpatB(v, half);
                bfrag[ks][2 * v]     = (f16)w_qkv[n * DIMIN + kb];
                bfrag[ks][2 * v + 1] = (f16)w_qkv[n * DIMIN + kb + 1];
            }
        }
        const float bias = b_qkv[n];
#pragma unroll
        for (int mt = 0; mt < 4; ++mt) {
            v8f acc = {};
#pragma unroll
            for (int ks = 0; ks < 3; ++ks) {
                v16h a;
                int arow = mt * 16 + l16;
#pragma unroll
                for (int v = 0; v < 8; ++v) {
                    int ka = ks * 32 + kpatA(v, half);
                    a[2 * v]     = xs[arow * DIMIN + ka];
                    a[2 * v + 1] = xs[arow * DIMIN + ka + 1];
                }
                acc = __builtin_amdgcn_wmma_f32_16x16x32_f16(false, a, false, bfrag[ks],
                                                             (short)0, acc, false, false);
            }
#pragma unroll
            for (int v = 0; v < 8; ++v) {
                int t = mt * 16 + v + 8 * half;
                qkvs[t * QKV_C + n] = (f16)(acc[v] + bias);
            }
        }
    }
    __syncthreads();

    // ---- Q pooling: max over 4 stride groups (token = qs*16 + r), fold in 1/sqrt(96)
    {
        const float scale = 0.10206207261596575f;         // 96^-0.5
        for (int i = tid; i < NHEADS * WIN * HEAD_D; i += 256) {
            int h = i / (WIN * HEAD_D);
            int r = (i / HEAD_D) % WIN;
            int d = i % HEAD_D;
            float m = -3.0e38f;
#pragma unroll
            for (int qs = 0; qs < 4; ++qs) {
                float vq = (float)qkvs[(qs * 16 + r) * QKV_C + h * HEAD_D + d];
                m = fmaxf(m, vq);
            }
            qp[i] = (f16)(m * scale);
        }
    }
    __syncthreads();

    // ---- Phase C: S = (Q*scale) @ K^T   [16x96]x[96x64] per head; 8 tiles == 8 waves
    {
        int h = wv >> 2, nt = wv & 3;
        v8f acc = {};
#pragma unroll
        for (int ks = 0; ks < 3; ++ks) {
            v16h a, bm;
            int tok = nt * 16 + l16;                      // B lane = key token
#pragma unroll
            for (int v = 0; v < 8; ++v) {
                int ka = ks * 32 + kpatA(v, half);
                a[2 * v]     = qp[(h * WIN + l16) * HEAD_D + ka];
                a[2 * v + 1] = qp[(h * WIN + l16) * HEAD_D + ka + 1];
                int kb = ks * 32 + kpatB(v, half);
                bm[2 * v]     = qkvs[tok * QKV_C + DIM_OUT + h * HEAD_D + kb];
                bm[2 * v + 1] = qkvs[tok * QKV_C + DIM_OUT + h * HEAD_D + kb + 1];
            }
            acc = __builtin_amdgcn_wmma_f32_16x16x32_f16(false, a, false, bm,
                                                         (short)0, acc, false, false);
        }
#pragma unroll
        for (int v = 0; v < 8; ++v) {
            int m = v + 8 * half;
            S[(h * WIN + m) * TPW + nt * 16 + l16] = acc[v];
        }
    }
    __syncthreads();

    // ---- Phase D: softmax over the 64 keys; one thread per (head,row) = 32 rows
    if (tid < NHEADS * WIN) {
        float* row = S + tid * TPW;
        float mx = row[0];
        for (int i = 1; i < TPW; ++i) mx = fmaxf(mx, row[i]);
        float sum = 0.f;
        for (int i = 0; i < TPW; ++i) {
            float e = __expf(row[i] - mx);
            sum += e;
            P[tid * TPW + i] = (f16)e;
        }
        float inv = 1.0f / sum;
        for (int i = 0; i < TPW; ++i)
            P[tid * TPW + i] = (f16)((float)P[tid * TPW + i] * inv);
    }
    __syncthreads();

    // ---- Phase E: O = P @ V  [16x64]x[64x96] per head; 12 tiles over 8 waves
    for (int tile = wv; tile < 12; tile += 8) {
        int h = tile / 6, nt = tile % 6;
        v8f acc = {};
#pragma unroll
        for (int ks = 0; ks < 2; ++ks) {
            v16h a, bm;
            int d = nt * 16 + l16;                        // B lane = head-dim channel
#pragma unroll
            for (int v = 0; v < 8; ++v) {
                int ka = ks * 32 + kpatA(v, half);
                a[2 * v]     = P[(h * WIN + l16) * TPW + ka];
                a[2 * v + 1] = P[(h * WIN + l16) * TPW + ka + 1];
                int kb = ks * 32 + kpatB(v, half);        // K = value-token index
                bm[2 * v]     = qkvs[kb * QKV_C + 2 * DIM_OUT + h * HEAD_D + d];
                bm[2 * v + 1] = qkvs[(kb + 1) * QKV_C + 2 * DIM_OUT + h * HEAD_D + d];
            }
            acc = __builtin_amdgcn_wmma_f32_16x16x32_f16(false, a, false, bm,
                                                         (short)0, acc, false, false);
        }
#pragma unroll
        for (int v = 0; v < 8; ++v) {
            int m = v + 8 * half;
            Ot[m * DIM_OUT + h * HEAD_D + nt * 16 + l16] = (f16)acc[v];
        }
    }
    __syncthreads();

    // ---- Phase F: out = Ot[16x192] @ w_proj^T[192x192] + b_proj
    // A-fragments are nt-invariant: hoist all 6 out of the tile loop.
    {
        v16h afr[6];
#pragma unroll
        for (int ks = 0; ks < 6; ++ks) {
#pragma unroll
            for (int v = 0; v < 8; ++v) {
                int ka = ks * 32 + kpatA(v, half);
                afr[ks][2 * v]     = Ot[l16 * DIM_OUT + ka];
                afr[ks][2 * v + 1] = Ot[l16 * DIM_OUT + ka + 1];
            }
        }
        float* outb = out + (size_t)b * (WIN * NW) * DIM_OUT;
        for (int nt = wv; nt < 12; nt += 8) {
            const int n = nt * 16 + l16;                  // output channel == row of w_proj
            v8f acc = {};
#pragma unroll
            for (int ks = 0; ks < 6; ++ks) {
                v16h bm;
#pragma unroll
                for (int v = 0; v < 8; ++v) {
                    int kb = ks * 32 + kpatB(v, half);
                    bm[2 * v]     = (f16)w_proj[n * DIM_OUT + kb];
                    bm[2 * v + 1] = (f16)w_proj[n * DIM_OUT + kb + 1];
                }
                acc = __builtin_amdgcn_wmma_f32_16x16x32_f16(false, afr[ks], false, bm,
                                                             (short)0, acc, false, false);
            }
            const float bias = b_proj[n];
#pragma unroll
            for (int v = 0; v < 8; ++v) {
                int m = v + 8 * half;                     // pooled token index t
                outb[(size_t)(m * NW + wwin) * DIM_OUT + n] = acc[v] + bias;
            }
        }
    }
}

extern "C" void kernel_launch(void* const* d_in, const int* in_sizes, int n_in,
                              void* d_out, int out_size, void* d_ws, size_t ws_size,
                              hipStream_t stream) {
    const float* x      = (const float*)d_in[0];
    const float* w_qkv  = (const float*)d_in[1];
    const float* b_qkv  = (const float*)d_in[2];
    const float* w_proj = (const float*)d_in[3];
    const float* b_proj = (const float*)d_in[4];
    float* out = (float*)d_out;

    dim3 grid(NBATCH * NW);   // 3136 workgroups: one per (batch, window)
    dim3 block(256);          // 8 wave32 waves
    muattn_fused_kernel<<<grid, block, SMEM_BYTES, stream>>>(
        x, w_qkv, b_qkv, w_proj, b_proj, out);
}
